// AttentionBlock_7679401525873
// MI455X (gfx1250) — matrix operand order repaired and checked by reading, not compiled, not run on hardware
//
#include <hip/hip_runtime.h>
#include <hip/hip_bf16.h>

// ---------------------------------------------------------------------------
// Problem constants (from reference): x (8, 512, 32, 32), 32 groups, 8 heads,
// head_dim 64, hid = 512, qkv rows = 1536, seq = H*W = 1024.
// ---------------------------------------------------------------------------
#define BATCH 8
#define CCH   512
#define SEQ   1024          // H*W
#define NGRP  32
#define CPG   16            // channels per group
#define NH    8
#define HD    64
#define HID   512           // NH*HD
#define OQ    1536          // 3*HID
#define EPSV  1e-6f

typedef __attribute__((ext_vector_type(16))) __bf16          v16bf;
typedef __attribute__((ext_vector_type(8)))  float           v8f;
typedef __attribute__((ext_vector_type(8)))  unsigned short  v8u16;

// round-to-nearest-even f32 -> bf16 (as raw ushort)
__device__ __forceinline__ unsigned short bf16_rne(float x) {
    unsigned int u = __float_as_uint(x);
    u = (u + 0x7FFFu + ((u >> 16) & 1u)) >> 16;
    return (unsigned short)u;
}

union FragBF { v16bf bf; v8u16 h[2]; unsigned short u[16]; };

// A fragment (16x32 bf16, M x K), source row-major with leading dim lda.
// Layout: lane L holds row M = L&15; K = 16*(L>>4) .. +15 (contiguous,
// matching the B fragments) -> two contiguous 16-byte loads per lane.
__device__ __forceinline__ v16bf load_a_bf16(const unsigned short* base,
                                             int lda, int row0, int k0) {
    int lane = threadIdx.x & 31;
    int row  = row0 + (lane & 15);
    int kb   = k0 + ((lane >> 4) << 4);
    FragBF f;
    f.h[0] = *(const v8u16*)(base + (size_t)row * lda + kb);
    f.h[1] = *(const v8u16*)(base + (size_t)row * lda + kb + 8);
    return f.bf;
}

// B fragment (32x16 bf16, K x N) loaded from B^T storage (N rows, K
// contiguous, leading dim ldb). Contiguous-K layout (same as A): lane L holds column N = L&15,
// K = 16*(L>>4) .. +15 packed 2/VGPR across v0..7 -> 32 contiguous bytes.
__device__ __forceinline__ v16bf load_bT_bf16(const unsigned short* base,
                                              int ldb, int n0, int k0) {
    int lane = threadIdx.x & 31;
    int nn   = n0 + (lane & 15);
    int kb   = k0 + ((lane >> 4) << 4);
    FragBF f;
    f.h[0] = *(const v8u16*)(base + (size_t)nn * ldb + kb);
    f.h[1] = *(const v8u16*)(base + (size_t)nn * ldb + kb + 8);
    return f.bf;
}

__device__ __forceinline__ v8f wmma_bf16(v16bf a, v16bf b, v8f c) {
    return __builtin_amdgcn_wmma_f32_16x16x32_bf16(
        false, a, false, b, (short)0, c, false, false);
}

// ---------------------------------------------------------------------------
// Kernel 1: f32 -> bf16 cast (weights)
// ---------------------------------------------------------------------------
__global__ void k_cast_bf16(const float* __restrict__ in,
                            unsigned short* __restrict__ out, int n) {
    int i = blockIdx.x * blockDim.x + threadIdx.x;
    int stride = gridDim.x * blockDim.x;
    for (; i < n; i += stride) out[i] = bf16_rne(in[i]);
}

// ---------------------------------------------------------------------------
// Kernel 2: GroupNorm. One block per (b, group): 16 ch x 1024 px = 16384 f32.
// Writes normalized activations TRANSPOSED to h_t[b, s, c] as bf16 so the QKV
// GEMM sees a contiguous-K B^T layout.
// ---------------------------------------------------------------------------
__global__ __launch_bounds__(256)
void k_groupnorm(const float* __restrict__ x,
                 const float* __restrict__ gamma,
                 const float* __restrict__ beta,
                 unsigned short* __restrict__ h_t) {
    __shared__ float rs[256], rs2[256];
    int bg = blockIdx.x;
    int b  = bg >> 5;           // / NGRP
    int g  = bg & 31;
    const float* xp = x + ((size_t)b * CCH + (size_t)g * CPG) * SEQ;
    int tid = threadIdx.x;

    float s = 0.f, s2 = 0.f;
    for (int i = tid; i < CPG * SEQ; i += 256) {
        float v = xp[i];
        s += v; s2 += v * v;
    }
    rs[tid] = s; rs2[tid] = s2;
    __syncthreads();
    for (int off = 128; off > 0; off >>= 1) {
        if (tid < off) { rs[tid] += rs[tid + off]; rs2[tid] += rs2[tid + off]; }
        __syncthreads();
    }
    float mean = rs[0] * (1.0f / (CPG * SEQ));
    float var  = rs2[0] * (1.0f / (CPG * SEQ)) - mean * mean;
    float inv  = rsqrtf(var + EPSV);

    for (int i = tid; i < CPG * SEQ; i += 256) {
        int c  = i >> 10;       // / SEQ
        int sp = i & (SEQ - 1);
        int cg = g * CPG + c;
        float y = (xp[i] - mean) * inv * gamma[cg] + beta[cg];
        h_t[((size_t)b * SEQ + sp) * CCH + cg] = bf16_rne(y);
    }
}

// ---------------------------------------------------------------------------
// Kernel 3: QKV GEMM.  qkv[b,o,s] = w_in[o,:] . h[b,:,s] + b_in[o]
// M=1536, N=1024 (per b), K=512. Block tile 128x128, 8 waves of 64x32.
// Epilogue routes o -> (q|k|v, head, d); q is pre-scaled by 1/sqrt(HD).
// q,k -> (b,n,s,d) bf16 (contiguous d);  v -> (b,n,d,s) bf16 (contiguous s).
// ---------------------------------------------------------------------------
__global__ __launch_bounds__(256)
void k_qkv_gemm(const unsigned short* __restrict__ w_in_bf,
                const unsigned short* __restrict__ h_t,
                const float* __restrict__ b_in,
                unsigned short* __restrict__ q_buf,
                unsigned short* __restrict__ k_buf,
                unsigned short* __restrict__ v_buf) {
    int b  = blockIdx.z;
    int M0 = blockIdx.y * 128;
    int N0 = blockIdx.x * 128;
    int w  = threadIdx.x >> 5;
    int lane = threadIdx.x & 31;
    int m_off = (w & 1) * 64;
    int n_off = (w >> 1) * 32;

    const unsigned short* Bt = h_t + (size_t)b * SEQ * CCH; // rows s, ld CCH

    v8f acc[4][2];
#pragma unroll
    for (int i = 0; i < 4; ++i)
#pragma unroll
        for (int j = 0; j < 2; ++j) acc[i][j] = (v8f){0,0,0,0,0,0,0,0};

    for (int kk = 0; kk < CCH; kk += 32) {
        v16bf a[4], bb[2];
#pragma unroll
        for (int i = 0; i < 4; ++i)
            a[i] = load_a_bf16(w_in_bf, CCH, M0 + m_off + 16 * i, kk);
#pragma unroll
        for (int j = 0; j < 2; ++j)
            bb[j] = load_bT_bf16(Bt, CCH, N0 + n_off + 16 * j, kk);
#pragma unroll
        for (int i = 0; i < 4; ++i)
#pragma unroll
            for (int j = 0; j < 2; ++j)
                acc[i][j] = wmma_bf16(a[i], bb[j], acc[i][j]);
    }

#pragma unroll
    for (int i = 0; i < 4; ++i) {
#pragma unroll
        for (int j = 0; j < 2; ++j) {
            int o8 = M0 + m_off + 16 * i + ((lane >> 4) << 3); // 8 contig o
            int sp = N0 + n_off + 16 * j + (lane & 15);
            int qi   = o8 >> 9;          // 0:q 1:k 2:v
            int head = (o8 >> 6) & 7;
            int d0   = o8 & 63;
            float vals[8];
#pragma unroll
            for (int r = 0; r < 8; ++r) vals[r] = acc[i][j][r] + b_in[o8 + r];
            size_t bn = (size_t)b * NH + head;
            if (qi == 0) {
                v8u16 pk;
#pragma unroll
                for (int r = 0; r < 8; ++r) pk[r] = bf16_rne(vals[r] * 0.125f);
                *(v8u16*)(q_buf + (bn * SEQ + sp) * HD + d0) = pk;
            } else if (qi == 1) {
                v8u16 pk;
#pragma unroll
                for (int r = 0; r < 8; ++r) pk[r] = bf16_rne(vals[r]);
                *(v8u16*)(k_buf + (bn * SEQ + sp) * HD + d0) = pk;
            } else {
#pragma unroll
                for (int r = 0; r < 8; ++r)
                    v_buf[(bn * HD + d0 + r) * SEQ + sp] = bf16_rne(vals[r]);
            }
        }
    }
}

// ---------------------------------------------------------------------------
// Kernel 4: Attention for one (b, head) and a 16-row q tile.
// Phase 1: logits (16 x 1024) via WMMA -> 64 KB LDS (f32).
// Phase 2: row softmax in registers (shuffle reductions), normalized in place.
// Phase 3: out[c,q] = V (A, contiguous-K from global) x P^T (B from LDS,
//          16 contiguous f32 per lane -> cvt to bf16). Stored (b, s, hid) bf16.
// ---------------------------------------------------------------------------
__global__ __launch_bounds__(256)
void k_attention(const unsigned short* __restrict__ q_buf,
                 const unsigned short* __restrict__ k_buf,
                 const unsigned short* __restrict__ v_buf,
                 unsigned short* __restrict__ attn_t) {
    __shared__ float lg[16 * SEQ];           // 64 KB
    int blk  = blockIdx.x;
    int s0   = (blk & 63) << 4;              // q tile start
    int bn   = blk >> 6;                     // b*NH + n
    int w    = threadIdx.x >> 5;
    int lane = threadIdx.x & 31;

    const unsigned short* qp = q_buf + (size_t)bn * SEQ * HD;
    const unsigned short* kp = k_buf + (size_t)bn * SEQ * HD;
    const unsigned short* vp = v_buf + (size_t)bn * HD * SEQ;

    // ---- Phase 1: logits ----
    v16bf a0 = load_a_bf16(qp, HD, s0, 0);
    v16bf a1 = load_a_bf16(qp, HD, s0, 32);
#pragma unroll
    for (int j = 0; j < 8; ++j) {
        int kc = w * 128 + j * 16;
        v16bf b0 = load_bT_bf16(kp, HD, kc, 0);
        v16bf b1 = load_bT_bf16(kp, HD, kc, 32);
        v8f acc = (v8f){0,0,0,0,0,0,0,0};
        acc = wmma_bf16(a0, b0, acc);
        acc = wmma_bf16(a1, b1, acc);
#pragma unroll
        for (int r = 0; r < 8; ++r)
            lg[(r + ((lane >> 4) << 3)) * SEQ + kc + (lane & 15)] = acc[r];
    }
    __syncthreads();

    // ---- Phase 2: softmax (2 rows per wave, 32 elems per lane) ----
#pragma unroll
    for (int rr = 2 * w; rr < 2 * w + 2; ++rr) {
        float e[32];
        float m = -3.4e38f;
#pragma unroll
        for (int i = 0; i < 32; ++i) {
            e[i] = lg[rr * SEQ + i * 32 + lane];
            m = fmaxf(m, e[i]);
        }
#pragma unroll
        for (int off = 16; off > 0; off >>= 1)
            m = fmaxf(m, __shfl_xor(m, off, 32));
        float sum = 0.f;
#pragma unroll
        for (int i = 0; i < 32; ++i) { e[i] = __expf(e[i] - m); sum += e[i]; }
#pragma unroll
        for (int off = 16; off > 0; off >>= 1)
            sum += __shfl_xor(sum, off, 32);
        float inv = 1.0f / sum;
#pragma unroll
        for (int i = 0; i < 32; ++i) lg[rr * SEQ + i * 32 + lane] = e[i] * inv;
    }
    __syncthreads();

    // ---- Phase 3: out[c,q] = V . P^T (waves 0..3; M=16 c each, N=16 q) ----
    if (w < 4) {
        int c0 = w * 16;
        v8f acc = (v8f){0,0,0,0,0,0,0,0};
        for (int kk = 0; kk < SEQ; kk += 32) {
            v16bf av = load_a_bf16(vp, SEQ, c0, kk);     // V rows c, contig k
            // B fragment from LDS f32 probabilities: col q = lane&15,
            // K = kk + 16*(lane>>4) .. +15, converted to packed bf16.
            const float* p = &lg[(lane & 15) * SEQ + kk + ((lane >> 4) << 4)];
            FragBF fb;
#pragma unroll
            for (int t = 0; t < 16; ++t) fb.u[t] = bf16_rne(p[t]);
            acc = wmma_bf16(av, fb.bf, acc);
        }
        int sq = s0 + (lane & 15);
        int cb = c0 + ((lane >> 4) << 3);
        int b  = bn >> 3, n = bn & 7;
        v8u16 pk;
#pragma unroll
        for (int r = 0; r < 8; ++r) pk[r] = bf16_rne(acc[r]);
        *(v8u16*)(attn_t + ((size_t)(b * SEQ + sq)) * HID + n * HD + cb) = pk;
    }
}

// ---------------------------------------------------------------------------
// Kernel 5: output projection + bias + residual (f32 out).
// out[b,o,s] = w_out[o,:] . attn[b,:,s] + b_out[o] + x[b,o,s]
// M=512, N=1024 (per b), K=512. Same tiling as QKV GEMM.
// ---------------------------------------------------------------------------
__global__ __launch_bounds__(256)
void k_out_proj(const unsigned short* __restrict__ w_out_bf,
                const unsigned short* __restrict__ attn_t,
                const float* __restrict__ b_out,
                const float* __restrict__ x,
                float* __restrict__ out) {
    int b  = blockIdx.z;
    int M0 = blockIdx.y * 128;
    int N0 = blockIdx.x * 128;
    int w  = threadIdx.x >> 5;
    int lane = threadIdx.x & 31;
    int m_off = (w & 1) * 64;
    int n_off = (w >> 1) * 32;

    const unsigned short* Bt = attn_t + (size_t)b * SEQ * HID;

    v8f acc[4][2];
#pragma unroll
    for (int i = 0; i < 4; ++i)
#pragma unroll
        for (int j = 0; j < 2; ++j) acc[i][j] = (v8f){0,0,0,0,0,0,0,0};

    for (int kk = 0; kk < HID; kk += 32) {
        v16bf a[4], bb[2];
#pragma unroll
        for (int i = 0; i < 4; ++i)
            a[i] = load_a_bf16(w_out_bf, HID, M0 + m_off + 16 * i, kk);
#pragma unroll
        for (int j = 0; j < 2; ++j)
            bb[j] = load_bT_bf16(Bt, HID, N0 + n_off + 16 * j, kk);
#pragma unroll
        for (int i = 0; i < 4; ++i)
#pragma unroll
            for (int j = 0; j < 2; ++j)
                acc[i][j] = wmma_bf16(a[i], bb[j], acc[i][j]);
    }

#pragma unroll
    for (int i = 0; i < 4; ++i) {
#pragma unroll
        for (int j = 0; j < 2; ++j) {
            int o8 = M0 + m_off + 16 * i + ((lane >> 4) << 3);
            int sp = N0 + n_off + 16 * j + (lane & 15);
#pragma unroll
            for (int r = 0; r < 8; ++r) {
                size_t idx = ((size_t)b * CCH + o8 + r) * SEQ + sp;
                out[idx] = acc[i][j][r] + b_out[o8 + r] + x[idx];
            }
        }
    }
}

// ---------------------------------------------------------------------------
// Host launcher
// ---------------------------------------------------------------------------
extern "C" void kernel_launch(void* const* d_in, const int* in_sizes, int n_in,
                              void* d_out, int out_size, void* d_ws, size_t ws_size,
                              hipStream_t stream) {
    const float* x     = (const float*)d_in[0];
    const float* gamma = (const float*)d_in[1];
    const float* beta  = (const float*)d_in[2];
    const float* w_in  = (const float*)d_in[3];
    const float* b_in  = (const float*)d_in[4];
    const float* w_out = (const float*)d_in[5];
    const float* b_out = (const float*)d_in[6];
    float* out = (float*)d_out;

    char* ws = (char*)d_ws;
    unsigned short* h_t      = (unsigned short*)ws; ws += (size_t)BATCH * SEQ * CCH * 2;
    unsigned short* w_in_bf  = (unsigned short*)ws; ws += (size_t)OQ * CCH * 2;
    unsigned short* w_out_bf = (unsigned short*)ws; ws += (size_t)CCH * HID * 2;
    unsigned short* q_buf    = (unsigned short*)ws; ws += (size_t)BATCH * NH * SEQ * HD * 2;
    unsigned short* k_buf    = (unsigned short*)ws; ws += (size_t)BATCH * NH * SEQ * HD * 2;
    unsigned short* v_buf    = (unsigned short*)ws; ws += (size_t)BATCH * NH * HD * SEQ * 2;
    unsigned short* attn_t   = (unsigned short*)ws; ws += (size_t)BATCH * SEQ * HID * 2;

    k_cast_bf16<<<1024, 256, 0, stream>>>(w_in,  w_in_bf,  OQ * CCH);
    k_cast_bf16<<<512,  256, 0, stream>>>(w_out, w_out_bf, CCH * HID);

    k_groupnorm<<<BATCH * NGRP, 256, 0, stream>>>(x, gamma, beta, h_t);

    k_qkv_gemm<<<dim3(SEQ / 128, OQ / 128, BATCH), 256, 0, stream>>>(
        w_in_bf, h_t, b_in, q_buf, k_buf, v_buf);

    k_attention<<<BATCH * NH * (SEQ / 16), 256, 0, stream>>>(
        q_buf, k_buf, v_buf, attn_t);

    k_out_proj<<<dim3(SEQ / 128, CCH / 128, BATCH), 256, 0, stream>>>(
        w_out_bf, attn_t, b_out, x, out);
}
